// JacobiKANLinear_55808805044638
// MI455X (gfx1250) — matrix operand order repaired and checked by reading, not compiled, not run on hardware
//
#include <hip/hip_runtime.h>
#include <hip/hip_bf16.h>
#include <cstdint>
#include <cstddef>

// ---------------------------------------------------------------------------
// JacobiKAN linear for MI455X (gfx1250, wave32, WMMA + async global->LDS).
//
// out = silu(x) @ Wb^T + einsum('bik,oik->bo', JacobiP(tanh(x)), C) + bias
//
// Fused as ONE bf16 GEMM with expanded K = 1024*7 = 7168:
//   A'[b, g*1024+i] = { silu(x[b,i]) (g=0),  P_{g-1}(tanh(x[b,i])) (g=1..6) }
//   W'[o, g*1024+i] = { Wb[o,i]      (g=0),  C[o,i,g-1]            (g=1..6) }
//   out = A' @ W'^T + bias            (2*8192*7168*1024 = 120 GFLOP)
//
// A' (112 MB) + W' (14 MB) fit in 192 MB L2 -> GEMM re-reads hit L2; the
// 32 MB f32 output is written with non-temporal stores so it does not evict
// the L2-resident working set.
// Compute: v_wmma_f32_16x16x32_bf16. Staging: GLOBAL_LOAD_ASYNC_TO_LDS_B128
// with double-buffered LDS (ASYNCcnt-ordered done signals).
// ---------------------------------------------------------------------------

#define TOKENS 8192
#define IN_F   1024
#define OUT_F  1024
#define DEGREE 5
#define KP     (IN_F * (DEGREE + 2))   // 7168 expanded-K

#define BM 128      // block tile M (tokens)
#define BN 128      // block tile N (outputs)
#define BK 32       // K step = one WMMA K
#define LDA 40      // padded LDS row stride (bf16 elems)
#define LDB 40

typedef __attribute__((ext_vector_type(16))) __bf16 v16bf;
typedef __attribute__((ext_vector_type(8)))  __bf16 v8bf;
typedef __attribute__((ext_vector_type(8)))  float  v8f;

// ---------------------------------------------------------------------------
// Kernel 1: expand activations -> bf16 A' [TOKENS][KP], K blocked by group g.
// ---------------------------------------------------------------------------
__global__ __launch_bounds__(256) void expand_x_kernel(
    const float* __restrict__ x, __bf16* __restrict__ Ap) {
  int idx = blockIdx.x * blockDim.x + threadIdx.x;
  if (idx >= TOKENS * IN_F) return;
  int b = idx / IN_F;
  int i = idx - b * IN_F;

  float xv = __builtin_nontemporal_load(x + idx);   // x is read exactly once
  float s  = xv / (1.0f + __expf(-xv));             // silu
  float t  = tanhf(xv);

  // Jacobi recurrence with A=B=1 (constants fold at compile time).
  float p[DEGREE + 1];
  p[0] = 1.0f;
  p[1] = 2.0f * t;
#pragma unroll
  for (int n = 2; n <= DEGREE; ++n) {
    float k     = (float)(n - 1);
    float alpha = 2.0f * k * (k + 2.0f) * (2.0f * k);
    float beta  = 0.0f;
    float gamma = (2.0f * k) * (2.0f * k + 1.0f) * (2.0f * k + 2.0f);
    float delta = 2.0f * k * k * (2.0f * k + 2.0f);
    p[n] = ((beta + alpha * t) / gamma) * p[n - 1] - (delta / gamma) * p[n - 2];
  }

  __bf16* row = Ap + (size_t)b * KP + i;   // coalesced 2B stores per group
  row[0] = (__bf16)s;
#pragma unroll
  for (int g = 0; g <= DEGREE; ++g) row[(size_t)(g + 1) * IN_F] = (__bf16)p[g];
}

// ---------------------------------------------------------------------------
// Kernel 2: pack weights -> bf16 W' [OUT_F][KP], same K layout as A'.
// ---------------------------------------------------------------------------
__global__ __launch_bounds__(256) void pack_w_kernel(
    const float* __restrict__ Wb, const float* __restrict__ C,
    __bf16* __restrict__ Wp) {
  int idx = blockIdx.x * blockDim.x + threadIdx.x;
  if (idx >= OUT_F * IN_F) return;
  int o = idx / IN_F;
  int i = idx - o * IN_F;

  __bf16* row = Wp + (size_t)o * KP + i;
  row[0] = (__bf16)Wb[idx];
  const float* c = C + (size_t)idx * (DEGREE + 1);
#pragma unroll
  for (int k = 0; k <= DEGREE; ++k) row[(size_t)(k + 1) * IN_F] = (__bf16)c[k];
}

// ---------------------------------------------------------------------------
// Kernel 3: bf16 WMMA GEMM: out[8192][1024] = A' W'^T + bias.
// 256 threads = 8 waves; block tile 128x128; wave tile 32x64 (2x4 WMMA).
// Double-buffered LDS fed by GLOBAL_LOAD_ASYNC_TO_LDS_B128 (ASYNCcnt).
// Manual two-stage unroll: buffer indices are compile-time constants so the
// D->C accumulator chain register-allocates in place (no phi copies).
// ---------------------------------------------------------------------------
__global__ __launch_bounds__(256, 2) void wmma_gemm_kernel(
    const __bf16* __restrict__ Ap, const __bf16* __restrict__ Wp,
    const float* __restrict__ bias, float* __restrict__ out) {
  __shared__ __bf16 As[2 * BM * LDA];   // 2 x 10 KB
  __shared__ __bf16 Bs[2 * BN * LDB];   // 2 x 10 KB

  const int tid  = threadIdx.x;
  const int lane = tid & 31;
  const int wave = tid >> 5;
  const int wm   = (wave & 3) * 32;        // wave M offset in block tile
  const int wn   = (wave >> 2) * 64;       // wave N offset in block tile
  const int half = lane >> 4;              // K-split lane half per ISA layout
  const int l16  = lane & 15;

  const int blockN = blockIdx.x * BN;
  const int blockM = blockIdx.y * BM;

  // Async staging: thread copies 16 bf16 (2 x B128) of A and of B per K-step.
  const int a_row = tid >> 1;              // 0..127
  const int a_k   = (tid & 1) * 16;        // 0 / 16
  const size_t a_base = (size_t)(blockM + a_row) * KP + a_k;
  const size_t b_base = (size_t)(blockN + a_row) * KP + a_k;

  // Per-thread LDS destination byte addresses (low 32 bits of generic ptr).
  uint32_t laA[2], laB[2];
#pragma unroll
  for (int b = 0; b < 2; ++b) {
    laA[b] = (uint32_t)(uintptr_t)&As[b * BM * LDA + a_row * LDA + a_k];
    laB[b] = (uint32_t)(uintptr_t)&Bs[b * BN * LDB + a_row * LDB + a_k];
  }

  v8f acc[2][4] = {};

  // Issue one tile's async copies (4 x B128 per thread) into LDS buffer `nb`.
  auto issue = [&](int ktElems, int nb) {
    uint64_t gaA = (uint64_t)(uintptr_t)(Ap + a_base + (size_t)ktElems);
    uint64_t gaB = (uint64_t)(uintptr_t)(Wp + b_base + (size_t)ktElems);
    asm volatile(
        "global_load_async_to_lds_b128 %0, %2, off\n\t"
        "global_load_async_to_lds_b128 %0, %2, off offset:16\n\t"
        "global_load_async_to_lds_b128 %1, %3, off\n\t"
        "global_load_async_to_lds_b128 %1, %3, off offset:16"
        :: "v"(laA[nb]), "v"(laB[nb]), "v"(gaA), "v"(gaB) : "memory");
  };

  // Compute one K-step from LDS buffer `cb` (compile-time constant).
  auto stage = [&](int cb) {
    const __bf16* Asb = As + cb * BM * LDA;
    const __bf16* Bsb = Bs + cb * BN * LDB;
    // A fragment 16x32 bf16 per ISA layout:
    //   lanes 0-15  (half=0): K{0..7} then K{16..23}
    //   lanes 16-31 (half=1): K{8..15} then K{24..31}
    union { v16bf v; v8bf h[2]; } afrag[2], bfrag[4];
#pragma unroll
    for (int ti = 0; ti < 2; ++ti) {
      const __bf16* ar = &Asb[(wm + ti * 16 + l16) * LDA];
      afrag[ti].h[0] = *(const v8bf*)(ar + half * 8);
      afrag[ti].h[1] = *(const v8bf*)(ar + half * 8 + 16);
    }
    // B fragment 32x16 bf16: lane = column; half 0 -> K{0..7},{16..23}, half 1 -> K{8..15},{24..31}.
#pragma unroll
    for (int tj = 0; tj < 4; ++tj) {
      const __bf16* br = &Bsb[(wn + tj * 16 + l16) * LDB];
      bfrag[tj].h[0] = *(const v8bf*)(br + half * 8);
      bfrag[tj].h[1] = *(const v8bf*)(br + half * 8 + 16);
    }
#pragma unroll
    for (int ti = 0; ti < 2; ++ti)
#pragma unroll
      for (int tj = 0; tj < 4; ++tj)
        acc[ti][tj] = __builtin_amdgcn_wmma_f32_16x16x32_bf16(
            false, afrag[ti].v, false, bfrag[tj].v,
            (short)0, acc[ti][tj], false, false);
  };

  const int NT = KP / BK;                  // 224 K-steps (even)
  issue(0, 0);                             // prologue: tile 0 -> buffer 0

  for (int kt = 0; kt < NT; kt += 2) {
    // ---- stage 0: compute tile kt from buf0, prefetch kt+1 -> buf1 ----
    __syncthreads();                       // buf1 free (readers of kt-1 done)
    issue((kt + 1) * BK, 1);
    asm volatile("s_wait_asynccnt 0x4" ::: "memory");  // tile kt landed
    __syncthreads();
    stage(0);

    // ---- stage 1: compute tile kt+1 from buf1, prefetch kt+2 -> buf0 ----
    __syncthreads();                       // buf0 free
    if (kt + 2 < NT) {
      issue((kt + 2) * BK, 0);
      asm volatile("s_wait_asynccnt 0x4" ::: "memory");
    } else {
      asm volatile("s_wait_asynccnt 0x0" ::: "memory");
    }
    __syncthreads();
    stage(1);
  }

  // Epilogue: C layout = VGPR v -> M = v + half*8, N = l16. Add bias.
  // Non-temporal stores: output is write-once; keep A'/W' resident in L2.
#pragma unroll
  for (int ti = 0; ti < 2; ++ti) {
#pragma unroll
    for (int tj = 0; tj < 4; ++tj) {
      int col = blockN + wn + tj * 16 + l16;
      float bv = bias[col];
#pragma unroll
      for (int v = 0; v < 8; ++v) {
        int row = blockM + wm + ti * 16 + half * 8 + v;
        __builtin_nontemporal_store(acc[ti][tj][v] + bv,
                                    &out[(size_t)row * OUT_F + col]);
      }
    }
  }
}

// ---------------------------------------------------------------------------
extern "C" void kernel_launch(void* const* d_in, const int* in_sizes, int n_in,
                              void* d_out, int out_size, void* d_ws, size_t ws_size,
                              hipStream_t stream) {
  const float* x    = (const float*)d_in[0];  // [8192,1024]
  const float* Wb   = (const float*)d_in[1];  // [1024,1024]
  const float* C    = (const float*)d_in[2];  // [1024,1024,6]
  const float* bias = (const float*)d_in[3];  // [1024]
  float* out = (float*)d_out;                 // [8192,1024]

  // Workspace: A' (TOKENS*KP bf16 = 112 MB) then W' (OUT_F*KP bf16 = 14 MB).
  __bf16* Ap = (__bf16*)d_ws;
  __bf16* Wp = (__bf16*)((char*)d_ws + (size_t)TOKENS * KP * sizeof(__bf16));

  expand_x_kernel<<<(TOKENS * IN_F) / 256, 256, 0, stream>>>(x, Ap);
  pack_w_kernel<<<(OUT_F * IN_F) / 256, 256, 0, stream>>>(Wb, C, Wp);

  dim3 grid(OUT_F / BN, TOKENS / BM);   // 8 x 64 = 512 blocks
  wmma_gemm_kernel<<<grid, 256, 0, stream>>>(Ap, Wp, bias, out);
}
